// classifier_network5_24086176596019
// MI455X (gfx1250) — compile-verified, matrix operand order checked by reading, not run on hardware
//
#include <hip/hip_runtime.h>
#include <math.h>

typedef __bf16 bf16_t;
typedef __attribute__((ext_vector_type(16))) __bf16        v16bf;
typedef __attribute__((ext_vector_type(8)))  __bf16        v8bf;
typedef __attribute__((ext_vector_type(8)))  float         v8f;
typedef __attribute__((ext_vector_type(4)))  float         v4f;
typedef __attribute__((ext_vector_type(4)))  unsigned int  v4u;
typedef __attribute__((ext_vector_type(8)))  int           v8i;
typedef __attribute__((ext_vector_type(4)))  int           v4i;

#define NROWS 8192
#define DIM   256
#define INV_TEMP 20.0f          // 1 / 0.05

// LDS tile layout: 128 rows x 512B data, padded to 528B pitch so that
// ds_load_b128 fragment reads rotate 4 banks per row (2-cycle floor, no
// 16-way conflicts).  A tile at offset 0, B tile at TILE_LDS.
#define LDS_PITCH 528
#define TILE_LDS  (128 * LDS_PITCH)            // 67584 B
#define RED_OFF   (2 * TILE_LDS)               // reduction arrays
#define LDS_TOTAL (RED_OFF + 4 * 128 * 4)      // 137216 B

union ABfrag { v8bf h[2]; v16bf v; };

// ---------------------------------------------------------------------------
// Kernel 1: row-normalize fp32 -> bf16.  One wave per row, 8 rows per block.
// ---------------------------------------------------------------------------
__global__ __launch_bounds__(256) void normalize_bf16_kernel(
    const float* __restrict__ cap, const float* __restrict__ img,
    bf16_t* __restrict__ capb, bf16_t* __restrict__ imgb)
{
    const int wave = threadIdx.x >> 5;
    const int lane = threadIdx.x & 31;
    const int row  = blockIdx.x * 8 + wave;
    const float* src = (blockIdx.y == 0) ? cap  : img;
    bf16_t*      dst = (blockIdx.y == 0) ? capb : imgb;

    const float* p = src + (size_t)row * DIM + lane * 8;
    v4f x0 = *(const v4f*)(p);
    v4f x1 = *(const v4f*)(p + 4);

    float ss = 0.f;
#pragma unroll
    for (int k = 0; k < 4; ++k) ss += x0[k] * x0[k] + x1[k] * x1[k];
#pragma unroll
    for (int off = 16; off > 0; off >>= 1) ss += __shfl_xor(ss, off, 32);

    const float inv = 1.0f / fmaxf(sqrtf(ss), 1e-8f);

    v8bf o;
#pragma unroll
    for (int k = 0; k < 4; ++k) {
        o[k]     = (bf16_t)(x0[k] * inv);
        o[k + 4] = (bf16_t)(x1[k] * inv);
    }
    *(v8bf*)(dst + (size_t)row * DIM + lane * 8) = o;
}

// ---------------------------------------------------------------------------
// TDM descriptor issue: load one 128-row x 512B tile into LDS at ldsOff,
// padding 16B after every 512B so the LDS pitch is 528B.
// D# layout per CDNA5 ISA 8.3 (group0) / 8.4 (group1).
// 6-arg builtin variant: (g0:u32x4, g1:i32x8, g2:i32x4, g3:i32x4, i32x8, cpol)
// ---------------------------------------------------------------------------
#if __has_builtin(__builtin_amdgcn_tensor_load_to_lds) && __has_builtin(__builtin_amdgcn_s_wait_tensorcnt)
#define USE_TDM 1
__device__ __forceinline__ void tdm_load_tile(unsigned long long gaddr,
                                              unsigned int ldsOff)
{
    v4u g0; v8i g1;
    v4i z4 = {0, 0, 0, 0};
    v8i z8 = {0, 0, 0, 0, 0, 0, 0, 0};
    // group0: count=1 | lds_addr | global_addr[56:0] | type=2
    g0[0] = 1u;
    g0[1] = ldsOff;
    g0[2] = (unsigned int)gaddr;
    g0[3] = (unsigned int)((gaddr >> 32) & 0x1FFFFFFu) | (2u << 30);
    // group1: data_size=3 (8B units), pad_enable, pad_interval=6 (512B),
    //         pad_amount=3 (16B)
    g1[0] = (int)((3u << 16) | (1u << 20) | (6u << 22) | (3u << 25));
    g1[1] = (int)(64u << 16);     // tensor_dim0 = 64 (8B units = 512B row)
    g1[2] = (int)(8192u << 16);   // tensor_dim1 = 8192 rows (low 16 bits)
    g1[3] = (int)(64u << 16);     // tile_dim0 = 64 (full row)
    g1[4] = (int)128u;            // tile_dim1 = 128 rows, tile_dim2 = 0
    g1[5] = (int)64u;             // tensor_dim0_stride = 64 (8B units)
    g1[6] = 0;                    // stride0 hi / stride1 lo
    g1[7] = 0;                    // stride1 hi
    __builtin_amdgcn_tensor_load_to_lds(g0, g1, z4, z4, z8, 0);
}
#else
#define USE_TDM 0
#endif

// ---------------------------------------------------------------------------
// Kernel 2: LDS-staged bf16 WMMA GEMM fused with exp + masked row/col sums.
// Block = 256 threads = 8 waves, one 128x128 tile of S per block.
// ---------------------------------------------------------------------------
__global__ __launch_bounds__(256) void gemm_lse_kernel(
    const bf16_t* __restrict__ capb, const bf16_t* __restrict__ imgb,
    const int* __restrict__ idx,
    float* __restrict__ rowAll, float* __restrict__ rowPos,
    float* __restrict__ colAll, float* __restrict__ colPos)
{
    extern __shared__ unsigned char smem[];
    float* sRowAll = (float*)(smem + RED_OFF);
    float* sRowPos = sRowAll + 128;
    float* sColAll = sRowPos + 128;
    float* sColPos = sColAll + 128;

    const int tid  = threadIdx.x;
    const int wave = tid >> 5;
    const int lane = tid & 31;
    const int l16  = lane & 15;
    const int hi   = lane >> 4;

    const int rowBase = blockIdx.y * 128 + wave * 16;
    const int colBase = blockIdx.x * 128;

    if (tid < 128) {
        sRowAll[tid] = 0.f; sRowPos[tid] = 0.f;
        sColAll[tid] = 0.f; sColPos[tid] = 0.f;
    }

    // ---- stage A (cap rows) and B (img rows) tiles into LDS ----
#if USE_TDM
    if (wave == 0) {
        tdm_load_tile((unsigned long long)(uintptr_t)
                          (capb + (size_t)blockIdx.y * 128 * DIM), 0u);
        tdm_load_tile((unsigned long long)(uintptr_t)
                          (imgb + (size_t)blockIdx.x * 128 * DIM),
                      (unsigned int)TILE_LDS);
        __builtin_amdgcn_s_wait_tensorcnt(0);
    }
#else
    {
        const int r    = tid >> 1;          // 128 rows, 2 threads each
        const int half = tid & 1;           // 256B halves
#pragma unroll
        for (int which = 0; which < 2; ++which) {
            const bf16_t* g = (which == 0)
                ? capb + (size_t)(blockIdx.y * 128 + r) * DIM
                : imgb + (size_t)(blockIdx.x * 128 + r) * DIM;
            const unsigned char* gp = (const unsigned char*)g + half * 256;
            unsigned char* lp = smem + which * TILE_LDS
                              + (size_t)r * LDS_PITCH + half * 256;
#pragma unroll
            for (int i = 0; i < 16; ++i)
                *(v4u*)(lp + i * 16) = *(const v4u*)(gp + i * 16);
        }
    }
#endif
    __syncthreads();

    // ---- WMMA main loop, fragments read from LDS ----
    const v8f zero = {0.f, 0.f, 0.f, 0.f, 0.f, 0.f, 0.f, 0.f};
    v8f acc[8];
#pragma unroll
    for (int t = 0; t < 8; ++t) acc[t] = zero;

    const unsigned char* aBase = smem + (size_t)(wave * 16 + l16) * LDS_PITCH
                               + hi * 16;
    const unsigned char* bBase = smem + TILE_LDS + (size_t)l16 * LDS_PITCH
                               + hi * 16;

#pragma unroll
    for (int kk = 0; kk < DIM; kk += 32) {
        ABfrag a;
        a.h[0] = *(const v8bf*)(aBase + kk * 2);        // K = kk+hi*8 ..
        a.h[1] = *(const v8bf*)(aBase + kk * 2 + 32);   // K = kk+16+hi*8 ..
#pragma unroll
        for (int t = 0; t < 8; ++t) {
            const unsigned char* bp = bBase + t * (16 * LDS_PITCH) + kk * 2;
            ABfrag b;
            b.h[0] = *(const v8bf*)(bp);
            b.h[1] = *(const v8bf*)(bp + 32);
            acc[t] = __builtin_amdgcn_wmma_f32_16x16x32_bf16(
                false, a.v, false, b.v, (short)0, acc[t], false, false);
        }
    }

    // ---- epilogue: e = exp(s/T - 1/T); fixed shift cancels in log-ratio.
    // acc[t][r] = S[rowBase + hi*8 + r, colBase + 16*t + l16]
    const int m0 = rowBase + hi * 8;
    int idxRow[8];
#pragma unroll
    for (int r = 0; r < 8; ++r) idxRow[r] = idx[m0 + r];

    float rAll[8] = {0.f, 0.f, 0.f, 0.f, 0.f, 0.f, 0.f, 0.f};
    float rPos[8] = {0.f, 0.f, 0.f, 0.f, 0.f, 0.f, 0.f, 0.f};

#pragma unroll
    for (int t = 0; t < 8; ++t) {
        const int col    = colBase + t * 16 + l16;
        const int idxCol = idx[col];
        float cAll = 0.f, cPos = 0.f;
#pragma unroll
        for (int r = 0; r < 8; ++r) {
            const float e = __expf(INV_TEMP * acc[t][r] - INV_TEMP);
            rAll[r] += e;
            cAll    += e;
            if (idxRow[r] == idxCol) { rPos[r] += e; cPos += e; }
        }
        atomicAdd(&sColAll[t * 16 + l16], cAll);
        atomicAdd(&sColPos[t * 16 + l16], cPos);
    }
#pragma unroll
    for (int r = 0; r < 8; ++r) {
        atomicAdd(&sRowAll[wave * 16 + hi * 8 + r], rAll[r]);
        atomicAdd(&sRowPos[wave * 16 + hi * 8 + r], rPos[r]);
    }
    __syncthreads();

    if (tid < 128) {
        atomicAdd(&rowAll[blockIdx.y * 128 + tid], sRowAll[tid]);
        atomicAdd(&rowPos[blockIdx.y * 128 + tid], sRowPos[tid]);
        atomicAdd(&colAll[blockIdx.x * 128 + tid], sColAll[tid]);
        atomicAdd(&colPos[blockIdx.x * 128 + tid], sColPos[tid]);
    }
}

// ---------------------------------------------------------------------------
// Kernel 3: loss = sum_i log(rowAll/rowPos) + sum_j log(colAll/colPos)
// ---------------------------------------------------------------------------
__global__ __launch_bounds__(256) void finalize_kernel(
    const float* __restrict__ rowAll, const float* __restrict__ rowPos,
    const float* __restrict__ colAll, const float* __restrict__ colPos,
    float* __restrict__ out)
{
    __shared__ float red[256];
    const int i = blockIdx.x * 256 + threadIdx.x;
    float v = (logf(rowAll[i]) - logf(rowPos[i]))
            + (logf(colAll[i]) - logf(colPos[i]));
    red[threadIdx.x] = v;
    __syncthreads();
#pragma unroll
    for (int s = 128; s > 0; s >>= 1) {
        if (threadIdx.x < s) red[threadIdx.x] += red[threadIdx.x + s];
        __syncthreads();
    }
    if (threadIdx.x == 0) atomicAdd(out, red[0]);
}

// ---------------------------------------------------------------------------
extern "C" void kernel_launch(void* const* d_in, const int* in_sizes, int n_in,
                              void* d_out, int out_size, void* d_ws, size_t ws_size,
                              hipStream_t stream)
{
    const float* cap = (const float*)d_in[0];
    const float* img = (const float*)d_in[1];
    const int*   idx = (const int*)d_in[2];
    float* out = (float*)d_out;

    char* ws = (char*)d_ws;
    const size_t matBytes = (size_t)NROWS * DIM * sizeof(bf16_t);   // 4 MB
    bf16_t* capb  = (bf16_t*)ws;
    bf16_t* imgb  = (bf16_t*)(ws + matBytes);
    float* rowAll = (float*)(ws + 2 * matBytes);
    float* rowPos = rowAll + NROWS;
    float* colAll = rowPos + NROWS;
    float* colPos = colAll + NROWS;

    (void)hipMemsetAsync(rowAll, 0, 4 * (size_t)NROWS * sizeof(float), stream);
    (void)hipMemsetAsync(out, 0, sizeof(float), stream);

    dim3 gN(NROWS / 8, 2);
    normalize_bf16_kernel<<<gN, 256, 0, stream>>>(cap, img, capb, imgb);

    dim3 gG(NROWS / 128, NROWS / 128);
    gemm_lse_kernel<<<gG, 256, LDS_TOTAL, stream>>>(capb, imgb, idx,
                                                    rowAll, rowPos,
                                                    colAll, colPos);

    finalize_kernel<<<NROWS / 256, 256, 0, stream>>>(rowAll, rowPos,
                                                     colAll, colPos, out);
}
